// CausalSelfAttention_78958678770142
// MI455X (gfx1250) — matrix operand order repaired and checked by reading, not compiled, not run on hardware
//
#include <hip/hip_runtime.h>
#include <hip/hip_bf16.h>
#include <math.h>

typedef __attribute__((ext_vector_type(16))) __bf16 v16bf;
typedef __attribute__((ext_vector_type(8)))  __bf16 v8bf;
typedef __attribute__((ext_vector_type(8)))  float  v8f;
typedef unsigned int u32;
typedef __attribute__((ext_vector_type(4))) u32 u32x4;
typedef __attribute__((ext_vector_type(8))) u32 u32x8;

#define N_HEADC 12
#define HS      64
#define TTOK    512
#define BBAT    4
#define CCH     768
#define TOPM    16
#define MAXS    8
#define EPSF    1e-6f

// ---------------------------------------------------------------------------
// TDM helpers (CDNA5 ISA).
// 2D bf16 tile load, rows of 32 elements (64B = 16 DWORDs), LDS-padded by
// 4 DWORDs per row -> 80B row stride (bank-conflict-free for ds_load_b128
// fragment reads, 20*r mod 64 hits 16 distinct banks).
// ---------------------------------------------------------------------------
__device__ __forceinline__ unsigned lds_off(const void* p) {
  // generic LDS pointers carry the shared aperture in addr[63:32];
  // addr[31:0] is the LDS byte offset (ISA 10.2 aperture table).
  return (unsigned)(unsigned long long)p;
}

__device__ __forceinline__ void tdm_load_2d_bf16(
    unsigned ldsaddr, const void* gptr,
    unsigned td0, unsigned td1,        // tensor dims (elements)
    unsigned tile0, unsigned tile1,    // tile dims  (elements)
    unsigned long long stride0)        // dim0 stride (elements)
{
  const unsigned long long ga = (unsigned long long)gptr;
  u32x4 g0;
  g0[0] = 1u;                                  // count=1, user descriptor
  g0[1] = ldsaddr;                             // lds_addr
  g0[2] = (u32)ga;                             // global_addr[31:0]
  g0[3] = (u32)((ga >> 32) & 0x01ffffffull)    // global_addr[56:32]
        | (2u << 30);                          // type = 2 ("image")
  u32x8 g1;
  g1[0] = (1u << 16)                           // data_size = 2 bytes
        | (1u << 20)                           // pad_enable
        | (3u << 22)                           // pad_interval: 16 DWORDs
        | (3u << 25);                          // pad_amount: 4 DWORDs
  g1[1] = (td0 & 0xffffu) << 16;               // tensor_dim0[15:0]
  g1[2] = ((td0 >> 16) & 0xffffu) | ((td1 & 0xffffu) << 16);
  g1[3] = ((td1 >> 16) & 0xffffu) | (tile0 << 16);
  g1[4] = tile1 & 0xffffu;                     // tile_dim1 (tile_dim2 = 0)
  g1[5] = (u32)stride0;                        // tensor_dim0_stride[31:0]
  g1[6] = (u32)((stride0 >> 32) & 0xffffull);  // stride[47:32] (dim1 stride=0)
  g1[7] = 0u;
  asm volatile("tensor_load_to_lds %0, %1" :: "s"(g0), "s"(g1) : "memory");
}

// ---------------------------------------------------------------------------
// One-shot prep: fp32 -> bf16 convert (x) and transposed convert (weights).
// ---------------------------------------------------------------------------
__global__ __launch_bounds__(256) void convert_bf16_kernel(
    const float* __restrict__ src, __bf16* __restrict__ dst)
{
  const int idx = (blockIdx.x * 256 + threadIdx.x) * 4;
  float4 f = *(const float4*)(src + idx);
  dst[idx + 0] = (__bf16)f.x;
  dst[idx + 1] = (__bf16)f.y;
  dst[idx + 2] = (__bf16)f.z;
  dst[idx + 3] = (__bf16)f.w;
}

// src is [K][N] fp32 row-major; dst is [N][K] bf16 row-major.
__global__ __launch_bounds__(256) void transpose_bf16_kernel(
    const float* __restrict__ src, __bf16* __restrict__ dst, int K, int N)
{
  __shared__ float tile[32][33];
  const int n0 = blockIdx.x * 32, k0 = blockIdx.y * 32;
  const int c = threadIdx.x & 31, r0 = threadIdx.x >> 5;
  #pragma unroll
  for (int qq = 0; qq < 4; ++qq) {
    const int r = r0 + qq * 8;
    tile[r][c] = src[(size_t)(k0 + r) * N + n0 + c];
  }
  __syncthreads();
  #pragma unroll
  for (int qq = 0; qq < 4; ++qq) {
    const int r = r0 + qq * 8;
    dst[(size_t)(n0 + r) * K + k0 + c] = (__bf16)tile[c][r];
  }
}

// ---------------------------------------------------------------------------
// Kernel 1: qkv = x @ W_attn + b_attn, scattered into q/k/v [B,H,T,64].
// M=2048, K=768, N=2304. Block tile 128x64, 8 waves, 2x2 WMMA per wave.
// A/B tiles streamed global->LDS by the TDM, double-buffered.
// ---------------------------------------------------------------------------
__global__ __launch_bounds__(256) void qkv_wmma_kernel(
    const __bf16* __restrict__ xb, const __bf16* __restrict__ WaT,
    const float* __restrict__ bias,
    float* __restrict__ qo, float* __restrict__ ko, float* __restrict__ vo)
{
  __shared__ alignas(16) __bf16 As[2][128][40];
  __shared__ alignas(16) __bf16 Bs[2][64][40];
  const int tid  = threadIdx.x;
  const int lane = tid & 31, wid = tid >> 5;
  const int bm = blockIdx.x * 128;
  const int bn = blockIdx.y * 64;
  const int wm = (wid & 3) * 32;
  const int wn = (wid >> 2) * 32;

  v8f acc[2][2];
  #pragma unroll
  for (int a = 0; a < 2; ++a)
    #pragma unroll
    for (int b = 0; b < 2; ++b)
      #pragma unroll
      for (int t = 0; t < 8; ++t) acc[a][b][t] = 0.f;

  if (wid == 0) { // prologue: stage K-slab 0 into buffer 0
    tdm_load_2d_bf16(lds_off(&As[0][0][0]), xb + (size_t)bm * CCH,
                     CCH, 2048, 32, 128, CCH);
    tdm_load_2d_bf16(lds_off(&Bs[0][0][0]), WaT + (size_t)bn * CCH,
                     CCH, 3 * CCH, 32, 64, CCH);
  }

  for (int it = 0; it < CCH / 32; ++it) {
    const int cur = it & 1, nxt = cur ^ 1;
    if (wid == 0) {
      if (it + 1 < CCH / 32) {
        const int k1 = (it + 1) * 32;
        tdm_load_2d_bf16(lds_off(&As[nxt][0][0]), xb + (size_t)bm * CCH + k1,
                         CCH, 2048, 32, 128, CCH);
        tdm_load_2d_bf16(lds_off(&Bs[nxt][0][0]), WaT + (size_t)bn * CCH + k1,
                         CCH, 3 * CCH, 32, 64, CCH);
        __builtin_amdgcn_s_wait_tensorcnt(2); // cur pair complete (in-order)
      } else {
        __builtin_amdgcn_s_wait_tensorcnt(0);
      }
    }
    __syncthreads();

    const int half = lane >> 4, r16 = lane & 15;
    v16bf afrag[2], bfrag[2];
    #pragma unroll
    for (int sm = 0; sm < 2; ++sm) { // A lane layout: K[8h..8h+7], K[16+8h..]
      const __bf16* p = &As[cur][wm + sm * 16 + r16][0];
      v8bf lo = *(const v8bf*)(p + 8 * half);
      v8bf hi = *(const v8bf*)(p + 16 + 8 * half);
      #pragma unroll
      for (int t = 0; t < 8; ++t) { afrag[sm][t] = lo[t]; afrag[sm][8 + t] = hi[t]; }
    }
    #pragma unroll
    for (int sn = 0; sn < 2; ++sn) { // B lane layout: K[8h..8h+7], K[16+8h..]
      const __bf16* p = &Bs[cur][wn + sn * 16 + r16][0];
      v8bf lo = *(const v8bf*)(p + 8 * half);
      v8bf hi = *(const v8bf*)(p + 16 + 8 * half);
      #pragma unroll
      for (int t = 0; t < 8; ++t) { bfrag[sn][t] = lo[t]; bfrag[sn][8 + t] = hi[t]; }
    }
    #pragma unroll
    for (int sm = 0; sm < 2; ++sm)
      #pragma unroll
      for (int sn = 0; sn < 2; ++sn)
        acc[sm][sn] = __builtin_amdgcn_wmma_f32_16x16x32_bf16(
            false, afrag[sm], false, bfrag[sn], (short)0, acc[sm][sn], false, false);
    __syncthreads();
  }

  // Epilogue: bias + scatter to q/k/v [B,H,T,64]
  const int half = lane >> 4, nn = lane & 15;
  #pragma unroll
  for (int sn = 0; sn < 2; ++sn) {
    const int nglob = bn + wn + sn * 16 + nn;
    const float bb = bias[nglob];
    const int which = nglob / CCH;
    const int nh = nglob - which * CCH;
    const int h = nh >> 6, col = nh & 63;
    float* dst = (which == 0) ? qo : ((which == 1) ? ko : vo);
    #pragma unroll
    for (int sm = 0; sm < 2; ++sm) {
      #pragma unroll
      for (int r = 0; r < 8; ++r) {
        const int m = bm + wm + sm * 16 + r + 8 * half;
        const int bidx = m >> 9, t = m & (TTOK - 1);
        dst[(((size_t)bidx * N_HEADC + h) * TTOK + t) * HS + col] = acc[sm][sn][r] + bb;
      }
    }
  }
}

// ---------------------------------------------------------------------------
// Kernel 2: greedy DPP subset selection, one wave32 per (b,h,i) token.
// Single-wave workgroups: barriers degrade to S_NOP. y written as bf16
// (it feeds the bf16 WMMA projection anyway).
// ---------------------------------------------------------------------------
__global__ __launch_bounds__(32) void dpp_select_kernel(
    const float* __restrict__ q, const float* __restrict__ k,
    const float* __restrict__ v, __bf16* __restrict__ y)
{
  const int g    = blockIdx.x;
  const int i    = g & (TTOK - 1);
  const int bh   = g >> 9;
  const int lane = threadIdx.x;
  const float* kb = k + (size_t)bh * TTOK * HS;
  const float* vb = v + (size_t)bh * TTOK * HS;
  const float* qi = q + ((size_t)bh * TTOK + i) * HS;
  const float* ki = kb + (size_t)i * HS;

  __shared__ float q_sh[HS];
  __shared__ float kc[TOPM][HS + 1];
  __shared__ float ksel[MAXS][HS + 1];
  __shared__ float G0[MAXS][MAXS];
  __shared__ int   selsh[MAXS];

  q_sh[lane * 2]     = qi[lane * 2];
  q_sh[lane * 2 + 1] = qi[lane * 2 + 1];
  const float k0a = ki[lane * 2], k0b = ki[lane * 2 + 1];
  ksel[0][lane * 2]     = k0a;
  ksel[0][lane * 2 + 1] = k0b;
  float kin = k0a * k0a + k0b * k0b;
  #pragma unroll
  for (int off = 16; off; off >>= 1) kin += __shfl_xor(kin, off, 32);
  if (lane == 0) { selsh[0] = i; G0[0][0] = kin; }
  __syncthreads();

  float simv[16];
  #pragma unroll
  for (int s = 0; s < 16; ++s) {
    const int j = s * 32 + lane;
    const float* kr = kb + (size_t)j * HS;
    float a = 0.f;
    #pragma unroll
    for (int d = 0; d < HS; d += 4) {
      float4 kk = *(const float4*)(kr + d);
      a += kk.x * q_sh[d] + kk.y * q_sh[d + 1] + kk.z * q_sh[d + 2] + kk.w * q_sh[d + 3];
    }
    simv[s] = (j <= i) ? a : -__builtin_inff();
  }

  // top-16 (ties -> smaller index): candidate r lives in lane r
  float myVal = -__builtin_inff();
  int   myIdx = 0;
  unsigned used = 0u;
  for (int r = 0; r < 16; ++r) {
    float bv = ((used >> 0) & 1u) ? -__builtin_inff() : simv[0];
    int   bj = lane;
    #pragma unroll
    for (int s = 1; s < 16; ++s) {
      const float vv = ((used >> s) & 1u) ? -__builtin_inff() : simv[s];
      if (vv > bv) { bv = vv; bj = s * 32 + lane; }
    }
    #pragma unroll
    for (int off = 16; off; off >>= 1) {
      const float ov = __shfl_xor(bv, off, 32);
      const int   oj = __shfl_xor(bj, off, 32);
      if (ov > bv || (ov == bv && oj < bj)) { bv = ov; bj = oj; }
    }
    if (lane == r) { myVal = bv; myIdx = bj; }
    if ((bj & 31) == lane) used |= 1u << (bj >> 5);
  }

  float cnorm = 0.f;
  bool  avail = false;
  if (lane < 16) {
    const float* kr = kb + (size_t)myIdx * HS;
    #pragma unroll
    for (int d = 0; d < HS; d += 4) {
      float4 kk = *(const float4*)(kr + d);
      kc[lane][d] = kk.x; kc[lane][d + 1] = kk.y;
      kc[lane][d + 2] = kk.z; kc[lane][d + 3] = kk.w;
      cnorm += kk.x * kk.x + kk.y * kk.y + kk.z * kk.z + kk.w * kk.w;
    }
    avail = (myVal > -__builtin_inff()) && (myIdx != i);
  }
  __syncthreads();

  int   count = 1;
  float cur_s = -logf(kin + EPSF);
  bool  done  = false;

  for (int it = 0; it < MAXS - 1; ++it) {
    float cross[8];
    #pragma unroll
    for (int j = 0; j < 8; ++j) {
      float a = 0.f;
      if (lane < 16 && j < count) {
        #pragma unroll
        for (int d = 0; d < HS; d += 4)
          a += kc[lane][d]     * ksel[j][d]     + kc[lane][d + 1] * ksel[j][d + 1]
             + kc[lane][d + 2] * ksel[j][d + 2] + kc[lane][d + 3] * ksel[j][d + 3];
      }
      cross[j] = a;
    }
    // identity-padded 8x8 Gram (insertion slot == count), in-register LU det
    float Am[8][8];
    #pragma unroll
    for (int a = 0; a < 8; ++a) {
      #pragma unroll
      for (int b = 0; b < 8; ++b) {
        float vv;
        if (a < count && b < count)        vv = G0[a][b];
        else if (a == count && b < count)  vv = cross[b];
        else if (b == count && a < count)  vv = cross[a];
        else if (a == count && b == count) vv = cnorm;
        else                               vv = (a == b) ? 1.f : 0.f;
        Am[a][b] = vv;
      }
    }
    float det = 1.f;
    #pragma unroll
    for (int p = 0; p < 8; ++p) {
      const float piv = Am[p][p];
      det *= piv;
      const float ip = 1.f / piv;
      #pragma unroll
      for (int r = p + 1; r < 8; ++r) {
        const float f = Am[r][p] * ip;
        #pragma unroll
        for (int c = p + 1; c < 8; ++c) Am[r][c] -= f * Am[p][c];
      }
    }
    float s = -logf(det + EPSF) / (float)(count + 1);
    if (!(lane < 16 && avail)) s = -__builtin_inff();

    float bs = s; int bc = lane;
    int av = (lane < 16 && avail) ? 1 : 0;
    #pragma unroll
    for (int off = 16; off; off >>= 1) {
      const float os = __shfl_xor(bs, off, 32);
      const int   oc = __shfl_xor(bc, off, 32);
      av |= __shfl_xor(av, off, 32);
      if (os > bs || (os == bs && oc < bc)) { bs = os; bc = oc; }
    }
    const bool accept = (!done) && (av != 0) && ((bs > cur_s) || (count < 2));

    const int   bestIdx = __shfl(myIdx, bc, 32);
    const float bestN   = __shfl(cnorm, bc, 32);
    float crossB[8];
    #pragma unroll
    for (int j = 0; j < 8; ++j) crossB[j] = __shfl(cross[j], bc, 32);

    if (accept) {
      if (lane == bc) avail = false;
      ksel[count][lane * 2]     = kc[bc][lane * 2];
      ksel[count][lane * 2 + 1] = kc[bc][lane * 2 + 1];
      if (lane == 0) {
        selsh[count] = bestIdx;
        #pragma unroll
        for (int j = 0; j < 8; ++j)
          if (j < count) { G0[count][j] = crossB[j]; G0[j][count] = crossB[j]; }
        G0[count][count] = bestN;
      }
      cur_s = bs;
      count += 1;
    } else {
      done = true;
    }
    __syncthreads();
  }

  float a0 = 0.f, a1 = 0.f;
  #pragma unroll
  for (int s2 = 0; s2 < MAXS; ++s2) {
    if (s2 < count) {
      const float* vr = vb + (size_t)selsh[s2] * HS;
      a0 += vr[lane * 2];
      a1 += vr[lane * 2 + 1];
    }
  }
  const float inv = 1.f / (float)count;
  __bf16* yo = y + ((size_t)bh * TTOK + i) * HS;
  yo[lane * 2]     = (__bf16)(a0 * inv);
  yo[lane * 2 + 1] = (__bf16)(a1 * inv);
}

// ---------------------------------------------------------------------------
// Kernel 3: out = y @ W_proj + b_proj. A streamed by TDM from [B,H,T,64] bf16.
// M=2048, K=768, N=768.
// ---------------------------------------------------------------------------
__global__ __launch_bounds__(256) void proj_wmma_kernel(
    const __bf16* __restrict__ yb, const __bf16* __restrict__ WpT,
    const float* __restrict__ bias, float* __restrict__ out)
{
  __shared__ alignas(16) __bf16 As[2][128][40];
  __shared__ alignas(16) __bf16 Bs[2][64][40];
  const int tid  = threadIdx.x;
  const int lane = tid & 31, wid = tid >> 5;
  const int bm = blockIdx.x * 128;
  const int bn = blockIdx.y * 64;
  const int wm = (wid & 3) * 32;
  const int wn = (wid >> 2) * 32;
  const int bidx = bm >> 9, t0 = bm & (TTOK - 1);

  v8f acc[2][2];
  #pragma unroll
  for (int a = 0; a < 2; ++a)
    #pragma unroll
    for (int b = 0; b < 2; ++b)
      #pragma unroll
      for (int t = 0; t < 8; ++t) acc[a][b][t] = 0.f;

  if (wid == 0) {
    // k0 = 0 -> head 0, d0 = 0
    const __bf16* asrc = yb + (((size_t)bidx * N_HEADC + 0) * TTOK + t0) * HS;
    tdm_load_2d_bf16(lds_off(&As[0][0][0]), asrc, HS, TTOK, 32, 128, HS);
    tdm_load_2d_bf16(lds_off(&Bs[0][0][0]), WpT + (size_t)bn * CCH,
                     CCH, CCH, 32, 64, CCH);
  }

  for (int it = 0; it < CCH / 32; ++it) {
    const int cur = it & 1, nxt = cur ^ 1;
    if (wid == 0) {
      if (it + 1 < CCH / 32) {
        const int k1 = (it + 1) * 32;
        const int h = k1 >> 6, d0 = k1 & 63;
        const __bf16* asrc =
            yb + (((size_t)bidx * N_HEADC + h) * TTOK + t0) * HS + d0;
        tdm_load_2d_bf16(lds_off(&As[nxt][0][0]), asrc, HS, TTOK, 32, 128, HS);
        tdm_load_2d_bf16(lds_off(&Bs[nxt][0][0]), WpT + (size_t)bn * CCH + k1,
                         CCH, CCH, 32, 64, CCH);
        __builtin_amdgcn_s_wait_tensorcnt(2);
      } else {
        __builtin_amdgcn_s_wait_tensorcnt(0);
      }
    }
    __syncthreads();

    const int half = lane >> 4, r16 = lane & 15;
    v16bf afrag[2], bfrag[2];
    #pragma unroll
    for (int sm = 0; sm < 2; ++sm) {
      const __bf16* p = &As[cur][wm + sm * 16 + r16][0];
      v8bf lo = *(const v8bf*)(p + 8 * half);
      v8bf hi = *(const v8bf*)(p + 16 + 8 * half);
      #pragma unroll
      for (int t = 0; t < 8; ++t) { afrag[sm][t] = lo[t]; afrag[sm][8 + t] = hi[t]; }
    }
    #pragma unroll
    for (int sn = 0; sn < 2; ++sn) {
      const __bf16* p = &Bs[cur][wn + sn * 16 + r16][0];
      v8bf lo = *(const v8bf*)(p + 8 * half);
      v8bf hi = *(const v8bf*)(p + 16 + 8 * half);
      #pragma unroll
      for (int t = 0; t < 8; ++t) { bfrag[sn][t] = lo[t]; bfrag[sn][8 + t] = hi[t]; }
    }
    #pragma unroll
    for (int sm = 0; sm < 2; ++sm)
      #pragma unroll
      for (int sn = 0; sn < 2; ++sn)
        acc[sm][sn] = __builtin_amdgcn_wmma_f32_16x16x32_bf16(
            false, afrag[sm], false, bfrag[sn], (short)0, acc[sm][sn], false, false);
    __syncthreads();
  }

  const int half = lane >> 4, nn = lane & 15;
  #pragma unroll
  for (int sn = 0; sn < 2; ++sn) {
    const int nglob = bn + wn + sn * 16 + nn;
    const float bb = bias[nglob];
    #pragma unroll
    for (int sm = 0; sm < 2; ++sm) {
      #pragma unroll
      for (int r = 0; r < 8; ++r) {
        const int m = bm + wm + sm * 16 + r + 8 * half;
        out[(size_t)m * CCH + nglob] = acc[sm][sn][r] + bb;
      }
    }
  }
}

// ---------------------------------------------------------------------------
extern "C" void kernel_launch(void* const* d_in, const int* in_sizes, int n_in,
                              void* d_out, int out_size, void* d_ws, size_t ws_size,
                              hipStream_t stream) {
  (void)in_sizes; (void)n_in; (void)out_size; (void)ws_size;
  const float* x      = (const float*)d_in[0];
  const float* W_attn = (const float*)d_in[1];
  const float* b_attn = (const float*)d_in[2];
  const float* W_proj = (const float*)d_in[3];
  const float* b_proj = (const float*)d_in[4];
  float* out = (float*)d_out;

  const size_t S = (size_t)BBAT * N_HEADC * TTOK * HS; // 1,572,864
  char* w = (char*)d_ws;
  float*  qw  = (float*)w;  w += S * 4;
  float*  kw  = (float*)w;  w += S * 4;
  float*  vw  = (float*)w;  w += S * 4;
  __bf16* ybf = (__bf16*)w; w += S * 2;
  __bf16* xbf = (__bf16*)w; w += (size_t)2048 * CCH * 2;
  __bf16* WaT = (__bf16*)w; w += (size_t)(3 * CCH) * CCH * 2;
  __bf16* WpT = (__bf16*)w;

  // one-shot precision/layout prep
  convert_bf16_kernel<<<dim3((2048 * CCH) / 1024), 256, 0, stream>>>(x, xbf);
  transpose_bf16_kernel<<<dim3((3 * CCH) / 32, CCH / 32), 256, 0, stream>>>(
      W_attn, WaT, CCH, 3 * CCH);
  transpose_bf16_kernel<<<dim3(CCH / 32, CCH / 32), 256, 0, stream>>>(
      W_proj, WpT, CCH, CCH);

  qkv_wmma_kernel<<<dim3(2048 / 128, (3 * CCH) / 64), 256, 0, stream>>>(
      xbf, WaT, b_attn, qw, kw, vw);
  dpp_select_kernel<<<dim3(BBAT * N_HEADC * TTOK), 32, 0, stream>>>(
      qw, kw, vw, ybf);
  proj_wmma_kernel<<<dim3(2048 / 128, CCH / 64), 256, 0, stream>>>(
      ybf, WpT, b_proj, out);
}
